// DensityFrequencyFusion_19628000542928
// MI455X (gfx1250) — matrix operand order repaired and checked by reading, not compiled, not run on hardware
//
#include <hip/hip_runtime.h>
#include <hip/hip_bf16.h>
#include <math.h>

typedef __bf16 bhalf;
typedef __attribute__((ext_vector_type(16))) __bf16 v16bf;
typedef __attribute__((ext_vector_type(4)))  __bf16 v4bf;
typedef __attribute__((ext_vector_type(8)))  float v8f;

#define BATCH 8
#define CIN   256
#define HIDC  128
#define DCH   384
#define NHEAD 4
#define HDIM  96
#define NTOK  1600   /* 40*40 */
#define AGENTS 49

/* GEMM tiling: block = 128 threads (4 waves), 64x64 tile, K-stage 64
 * (two WMMA K=32 steps per barrier). Each wave owns a 32x32 sub-tile.
 * Software pipeline: global->reg for stage k+1 issued before the WMMA block
 * of stage k, so the loadcnt wait lands a full stage later.
 * LDS layouts (both K-contiguous so fragments load as ds_load_b128):
 *   As[m][k]  stride TLD,   Bs[n][k]  stride TLD  */
#define BM 64
#define BN 64
#define BK 64
#define TLD 72   /* 72 halves = 144B = 9*16B -> fragment bases stay 16B aligned */

__device__ __forceinline__ v4bf cvt4(float4 v) {
  v4bf h;
  h[0] = (bhalf)v.x; h[1] = (bhalf)v.y; h[2] = (bhalf)v.z; h[3] = (bhalf)v.w;
  return h;
}

/* A fragment (16x32 bf16, MxK): lane holds row=lane&15,
 * K slots f[0..15] = koff+(lane>>4)*16+i */
__device__ __forceinline__ v16bf frag_a(const bhalf* As, int row0, int koff, int lane) {
  v16bf f;
  const bhalf* p = As + (row0 + (lane & 15)) * TLD + koff + ((lane >> 4) << 4);
#pragma unroll
  for (int i = 0; i < 16; ++i) f[i] = p[i];
  return f;
}
/* B fragment (32x16 bf16, KxN): lane holds col=lane&15, K = koff+(lane>>4)*16+i */
__device__ __forceinline__ v16bf frag_b(const bhalf* Bs, int col0, int koff, int lane) {
  v16bf f;
  const bhalf* p = Bs + (col0 + (lane & 15)) * TLD + koff + ((lane >> 4) << 4);
#pragma unroll
  for (int i = 0; i < 16; ++i) f[i] = p[i];
  return f;
}

#define WMMA_BF16(a, b, c) \
  __builtin_amdgcn_wmma_f32_16x16x32_bf16(false, a, false, b, (short)0, c, false, false)

/* Generic batched GEMM  C[z] = alpha * op(A[z]) * op(B[z])  (+ epilogue)
 * TA: A stored [K,M] (row stride lda).  TB: B stored [N,K] (row stride ldb).
 * EPI: 0 none, 1 +bias[m], 2 BN+SiLU (g[m]*acc+bb[m])*sigmoid, 3 +bias[m]+R. */
template<bool TA, bool TB, int EPI>
__global__ void __launch_bounds__(128)
gemm_wmma(const float* __restrict__ A, const float* __restrict__ B,
          float* __restrict__ C,
          int M, int N, int K, int lda, int ldb, int ldc,
          long batA, long batB, long batC,
          const float* __restrict__ g, const float* __restrict__ bb,
          const float* __restrict__ R, long batR, float alpha)
{
  __shared__ alignas(16) bhalf As[BM * TLD];
  __shared__ alignas(16) bhalf Bs[BN * TLD];
  const int z = blockIdx.z;
  A += (long)z * batA; B += (long)z * batB; C += (long)z * batC;
  const int m0 = blockIdx.y * BM, n0 = blockIdx.x * BN;
  const int tid = threadIdx.x, lane = tid & 31, wv = tid >> 5;
  const int wm = (wv >> 1) << 5, wn = (wv & 1) << 5;
  const bool fullM = (m0 + BM <= M), fullN = (n0 + BN <= N);
  const bool la4 = ((lda & 3) == 0), lb4 = ((ldb & 3) == 0);

  /* quad coordinates per thread (t=0..7): row = q>>4, quad-col = (q&15)*4 */
  int r8[8], c8[8];
#pragma unroll
  for (int t = 0; t < 8; ++t) {
    int q = t * 128 + tid;
    r8[t] = q >> 4;
    c8[t] = (q & 15) << 2;
  }

  float4 ra[8], rb[8];   /* staging registers (pipeline across K-stages) */

  auto g2rA = [&](int k0) {
    const bool fast = fullM && (k0 + BK <= K) && la4;
    if (TA) {            /* A stored [K,M]: row = k, quad over m */
      if (fast) {
#pragma unroll
        for (int t = 0; t < 8; ++t)
          ra[t] = *(const float4*)&A[(long)(k0 + r8[t]) * lda + m0 + c8[t]];
      } else {
#pragma unroll
        for (int t = 0; t < 8; ++t) {
          float* e = (float*)&ra[t];
          int gk = k0 + r8[t];
#pragma unroll
          for (int j = 0; j < 4; ++j) {
            int gm = m0 + c8[t] + j;
            bool ok = (gk < K) && (gm < M);
            long ix = ok ? ((long)gk * lda + gm) : 0;
            float v = A[ix];
            e[j] = ok ? v : 0.f;
          }
        }
      }
    } else {             /* A stored [M,K]: row = m, quad over k */
      if (fast) {
#pragma unroll
        for (int t = 0; t < 8; ++t)
          ra[t] = *(const float4*)&A[(long)(m0 + r8[t]) * lda + k0 + c8[t]];
      } else {
#pragma unroll
        for (int t = 0; t < 8; ++t) {
          float* e = (float*)&ra[t];
          int gm = m0 + r8[t];
#pragma unroll
          for (int j = 0; j < 4; ++j) {
            int gk = k0 + c8[t] + j;
            bool ok = (gm < M) && (gk < K);
            long ix = ok ? ((long)gm * lda + gk) : 0;
            float v = A[ix];
            e[j] = ok ? v : 0.f;
          }
        }
      }
    }
  };
  auto g2rB = [&](int k0) {
    const bool fast = fullN && (k0 + BK <= K) && lb4;
    if (TB) {            /* B stored [N,K]: row = n, quad over k */
      if (fast) {
#pragma unroll
        for (int t = 0; t < 8; ++t)
          rb[t] = *(const float4*)&B[(long)(n0 + r8[t]) * ldb + k0 + c8[t]];
      } else {
#pragma unroll
        for (int t = 0; t < 8; ++t) {
          float* e = (float*)&rb[t];
          int gn = n0 + r8[t];
#pragma unroll
          for (int j = 0; j < 4; ++j) {
            int gk = k0 + c8[t] + j;
            bool ok = (gn < N) && (gk < K);
            long ix = ok ? ((long)gn * ldb + gk) : 0;
            float v = B[ix];
            e[j] = ok ? v : 0.f;
          }
        }
      }
    } else {             /* B stored [K,N]: row = k, quad over n */
      if (fast) {
#pragma unroll
        for (int t = 0; t < 8; ++t)
          rb[t] = *(const float4*)&B[(long)(k0 + r8[t]) * ldb + n0 + c8[t]];
      } else {
#pragma unroll
        for (int t = 0; t < 8; ++t) {
          float* e = (float*)&rb[t];
          int gk = k0 + r8[t];
#pragma unroll
          for (int j = 0; j < 4; ++j) {
            int gn = n0 + c8[t] + j;
            bool ok = (gk < K) && (gn < N);
            long ix = ok ? ((long)gk * ldb + gn) : 0;
            float v = B[ix];
            e[j] = ok ? v : 0.f;
          }
        }
      }
    }
  };
  auto r2lds = [&]() {
    if (TA) {            /* transposed store into As[m][k] */
#pragma unroll
      for (int t = 0; t < 8; ++t) {
        v4bf h = cvt4(ra[t]);
        As[(c8[t] + 0) * TLD + r8[t]] = h[0];
        As[(c8[t] + 1) * TLD + r8[t]] = h[1];
        As[(c8[t] + 2) * TLD + r8[t]] = h[2];
        As[(c8[t] + 3) * TLD + r8[t]] = h[3];
      }
    } else {
#pragma unroll
      for (int t = 0; t < 8; ++t)
        *(v4bf*)&As[r8[t] * TLD + c8[t]] = cvt4(ra[t]);   /* ds_store_b64 */
    }
    if (TB) {
#pragma unroll
      for (int t = 0; t < 8; ++t)
        *(v4bf*)&Bs[r8[t] * TLD + c8[t]] = cvt4(rb[t]);   /* ds_store_b64 */
    } else {             /* transposed store into Bs[n][k] */
#pragma unroll
      for (int t = 0; t < 8; ++t) {
        v4bf h = cvt4(rb[t]);
        Bs[(c8[t] + 0) * TLD + r8[t]] = h[0];
        Bs[(c8[t] + 1) * TLD + r8[t]] = h[1];
        Bs[(c8[t] + 2) * TLD + r8[t]] = h[2];
        Bs[(c8[t] + 3) * TLD + r8[t]] = h[3];
      }
    }
  };

  v8f acc[2][2] = {};
  g2rA(0); g2rB(0);
  for (int k0 = 0; k0 < K; k0 += BK) {
    if (k0 + 2 * BK < K) { /* CDNA5 global_prefetch_b8, two stages ahead */
      __builtin_prefetch(TA ? (const void*)&A[(long)(k0 + 2 * BK) * lda + m0]
                            : (const void*)&A[(long)m0 * lda + k0 + 2 * BK], 0, 1);
      __builtin_prefetch(TB ? (const void*)&B[(long)n0 * ldb + k0 + 2 * BK]
                            : (const void*)&B[(long)(k0 + 2 * BK) * ldb + n0], 0, 1);
    }
    r2lds();               /* waits on in-flight loads from previous stage */
    __syncthreads();
    if (k0 + BK < K) { g2rA(k0 + BK); g2rB(k0 + BK); } /* overlap with WMMA */
#pragma unroll
    for (int kk = 0; kk < 2; ++kk) {
      v16bf a0 = frag_a(As, wm,      kk * 32, lane);
      v16bf a1 = frag_a(As, wm + 16, kk * 32, lane);
      v16bf b0 = frag_b(Bs, wn,      kk * 32, lane);
      v16bf b1 = frag_b(Bs, wn + 16, kk * 32, lane);
      acc[0][0] = WMMA_BF16(a0, b0, acc[0][0]);
      acc[0][1] = WMMA_BF16(a0, b1, acc[0][1]);
      acc[1][0] = WMMA_BF16(a1, b0, acc[1][0]);
      acc[1][1] = WMMA_BF16(a1, b1, acc[1][1]);
    }
    __syncthreads();
  }
  const int rbase = (lane < 16) ? 0 : 8;  /* C layout: VGPR i -> M = rbase+i */
  const int ccol = lane & 15;
#pragma unroll
  for (int ai = 0; ai < 2; ++ai)
#pragma unroll
    for (int bi = 0; bi < 2; ++bi)
#pragma unroll
      for (int i = 0; i < 8; ++i) {
        int m = m0 + wm + ai * 16 + rbase + i;
        int n = n0 + wn + bi * 16 + ccol;
        if (m < M && n < N) {
          float v = acc[ai][bi][i] * alpha;
          if (EPI == 1) v += bb[m];
          if (EPI == 2) { v = g[m] * v + bb[m]; v = v / (1.f + expf(-v)); }
          if (EPI == 3) v = v + bb[m] + R[(long)z * batR + (long)m * ldc + n];
          C[(long)m * ldc + n] = v;
        }
      }
}

/* p3: 3x3 stride-2 conv 256->128 (80x80 -> 40x40) as implicit-im2col WMMA GEMM.
 * M=128 (OC), N=1600 (oy,ox), K=2304 (ic,r,s); B gathered in the G2R phase. */
__global__ void __launch_bounds__(128)
gemm_p3_wmma(const float* __restrict__ W, const float* __restrict__ X,
             float* __restrict__ C, long batC,
             const float* __restrict__ g, const float* __restrict__ bb)
{
  const int M = 128, N = 1600, K = 2304;
  __shared__ alignas(16) bhalf As[BM * TLD];
  __shared__ alignas(16) bhalf Bs[BN * TLD];
  const int z = blockIdx.z;
  const float* Xb = X + (long)z * 256 * 6400;
  float* Cb = C + (long)z * batC;
  const int m0 = blockIdx.y * BM, n0 = blockIdx.x * BN;
  const int tid = threadIdx.x, lane = tid & 31, wv = tid >> 5;
  const int wm = (wv >> 1) << 5, wn = (wv & 1) << 5;

  int r8[8], c8[8];
#pragma unroll
  for (int t = 0; t < 8; ++t) {
    int q = t * 128 + tid;
    r8[t] = q >> 4;
    c8[t] = (q & 15) << 2;
  }
  float4 ra[8], rb[8];

  auto g2rA = [&](int k0) {
#pragma unroll
    for (int t = 0; t < 8; ++t)
      ra[t] = *(const float4*)&W[(long)(m0 + r8[t]) * K + k0 + c8[t]];
  };
  auto g2rB = [&](int k0) {   /* branchless im2col gather */
#pragma unroll
    for (int t = 0; t < 8; ++t) {
      float* e = (float*)&rb[t];
      int gk = k0 + r8[t];
      int ic = gk / 9, rs = gk % 9, rr = rs / 3, ss = rs % 3;
#pragma unroll
      for (int j = 0; j < 4; ++j) {
        int gn = n0 + c8[t] + j;
        int oy = gn / 40, ox = gn % 40;
        int iy = oy * 2 - 1 + rr, ix = ox * 2 - 1 + ss;
        bool ok = (gn < N) && (iy >= 0) && (iy < 80) && (ix >= 0) && (ix < 80);
        long ixg = ok ? ((long)ic * 6400 + iy * 80 + ix) : 0;
        float v = Xb[ixg];
        e[j] = ok ? v : 0.f;
      }
    }
  };
  auto r2lds = [&]() {
#pragma unroll
    for (int t = 0; t < 8; ++t)
      *(v4bf*)&As[r8[t] * TLD + c8[t]] = cvt4(ra[t]);
#pragma unroll
    for (int t = 0; t < 8; ++t) {
      v4bf h = cvt4(rb[t]);
      Bs[(c8[t] + 0) * TLD + r8[t]] = h[0];
      Bs[(c8[t] + 1) * TLD + r8[t]] = h[1];
      Bs[(c8[t] + 2) * TLD + r8[t]] = h[2];
      Bs[(c8[t] + 3) * TLD + r8[t]] = h[3];
    }
  };

  v8f acc[2][2] = {};
  g2rA(0); g2rB(0);
  for (int k0 = 0; k0 < K; k0 += BK) {
    r2lds();
    __syncthreads();
    if (k0 + BK < K) { g2rA(k0 + BK); g2rB(k0 + BK); }
#pragma unroll
    for (int kk = 0; kk < 2; ++kk) {
      v16bf a0 = frag_a(As, wm,      kk * 32, lane);
      v16bf a1 = frag_a(As, wm + 16, kk * 32, lane);
      v16bf b0 = frag_b(Bs, wn,      kk * 32, lane);
      v16bf b1 = frag_b(Bs, wn + 16, kk * 32, lane);
      acc[0][0] = WMMA_BF16(a0, b0, acc[0][0]);
      acc[0][1] = WMMA_BF16(a0, b1, acc[0][1]);
      acc[1][0] = WMMA_BF16(a1, b0, acc[1][0]);
      acc[1][1] = WMMA_BF16(a1, b1, acc[1][1]);
    }
    __syncthreads();
  }
  const int rbase = (lane < 16) ? 0 : 8;
  const int ccol = lane & 15;
#pragma unroll
  for (int ai = 0; ai < 2; ++ai)
#pragma unroll
    for (int bi = 0; bi < 2; ++bi)
#pragma unroll
      for (int i = 0; i < 8; ++i) {
        int m = m0 + wm + ai * 16 + rbase + i;
        int n = n0 + wn + bi * 16 + ccol;
        if (m < M && n < N) {
          float v = acc[ai][bi][i];
          v = g[m] * v + bb[m];
          v = v / (1.f + expf(-v));
          Cb[(long)m * 1600 + n] = v;
        }
      }
}

/* ---------------- elementwise / reduction glue ---------------- */

__global__ void resize_bilinear_p5(const float* __restrict__ src, float* __restrict__ dst) {
  long idx = (long)blockIdx.x * 256 + threadIdx.x;
  const long total = (long)BATCH * HIDC * NTOK;
  if (idx >= total) return;
  int n = (int)(idx % NTOK); long bc = idx / NTOK;
  int ox = n % 40, oy = n / 40;
  const float* p = src + bc * 400;
  float sy = (oy + 0.5f) * 0.5f - 0.5f, sx = (ox + 0.5f) * 0.5f - 0.5f;
  float fy = floorf(sy), fx = floorf(sx);
  float wy = sy - fy, wx = sx - fx;
  int y0 = (int)fy, x0 = (int)fx;
  int y1 = y0 + 1, x1 = x0 + 1;
  y0 = min(max(y0, 0), 19); y1 = min(max(y1, 0), 19);
  x0 = min(max(x0, 0), 19); x1 = min(max(x1, 0), 19);
  float v = (1.f - wy) * ((1.f - wx) * p[y0 * 20 + x0] + wx * p[y0 * 20 + x1])
          + wy        * ((1.f - wx) * p[y1 * 20 + x0] + wx * p[y1 * 20 + x1]);
  int b = (int)(bc / HIDC), c = (int)(bc % HIDC);
  dst[((long)b * DCH + c) * NTOK + n] = v;
}

/* Haar decomposition: xc[b,c,40,40] -> xw[b,c,4,20,20] */
__global__ void wav_dec_kernel(const float* __restrict__ xc, float* __restrict__ xw) {
  long idx = (long)blockIdx.x * 256 + threadIdx.x;
  const long total = (long)BATCH * DCH * 400;
  if (idx >= total) return;
  int ij = (int)(idx % 400); long bc = idx / 400;
  int i = ij / 20, j = ij % 20;
  const float* p = xc + bc * NTOK;
  float x00 = p[(2 * i) * 40 + 2 * j],     x01 = p[(2 * i) * 40 + 2 * j + 1];
  float x10 = p[(2 * i + 1) * 40 + 2 * j], x11 = p[(2 * i + 1) * 40 + 2 * j + 1];
  float* o = xw + bc * 1600 + ij;
  o[0]    = 0.5f * (x00 + x01 + x10 + x11);
  o[400]  = 0.5f * (x00 + x01 - x10 - x11);
  o[800]  = 0.5f * (x00 - x01 + x10 - x11);
  o[1200] = 0.5f * (x00 - x01 - x10 + x11);
}

/* depthwise 3x3 freq conv on [b,1536,20,20], xw2 = xw + fs*conv(xw) */
__global__ void wav_freq_kernel(const float* __restrict__ xw, const float* __restrict__ fw,
                                const float* __restrict__ fs, float* __restrict__ xw2) {
  long idx = (long)blockIdx.x * 256 + threadIdx.x;
  const long total = (long)BATCH * DCH * 4 * 400;
  if (idx >= total) return;
  int ij = (int)(idx % 400); long bcf = idx / 400;
  int cf = (int)(bcf % (DCH * 4));
  int i = ij / 20, j = ij % 20;
  const float* p = xw + bcf * 400;
  float acc = 0.f;
#pragma unroll
  for (int r = 0; r < 3; ++r)
#pragma unroll
    for (int s = 0; s < 3; ++s) {
      int ii = i - 1 + r, jj = j - 1 + s;
      if (ii >= 0 && ii < 20 && jj >= 0 && jj < 20)
        acc += p[ii * 20 + jj] * fw[cf * 9 + r * 3 + s];
    }
  xw2[idx] = p[ij] + fs[cf] * acc;
}

/* Haar reconstruction: xw2[b,c,4,20,20] -> xf[b,c,40,40] */
__global__ void wav_rec_kernel(const float* __restrict__ xw2, float* __restrict__ xf) {
  long idx = (long)blockIdx.x * 256 + threadIdx.x;
  const long total = (long)BATCH * DCH * 400;
  if (idx >= total) return;
  int ij = (int)(idx % 400); long bc = idx / 400;
  int i = ij / 20, j = ij % 20;
  const float* p = xw2 + bc * 1600 + ij;
  float f0 = p[0], f1 = p[400], f2 = p[800], f3 = p[1200];
  float* o = xf + bc * NTOK;
  o[(2 * i) * 40 + 2 * j]         = 0.5f * (f0 + f1 + f2 + f3);
  o[(2 * i) * 40 + 2 * j + 1]     = 0.5f * (f0 + f1 - f2 - f3);
  o[(2 * i + 1) * 40 + 2 * j]     = 0.5f * (f0 - f1 + f2 - f3);
  o[(2 * i + 1) * 40 + 2 * j + 1] = 0.5f * (f0 - f1 - f2 + f3);
}

/* adaptive avg pool 40x40 -> 7x7  :  agent[b][c][49] */
__global__ void agent_pool_kernel(const float* __restrict__ xf, float* __restrict__ agent) {
  long idx = (long)blockIdx.x * 256 + threadIdx.x;
  const long total = (long)BATCH * DCH * AGENTS;
  if (idx >= total) return;
  int a = (int)(idx % AGENTS); long bc = idx / AGENTS;
  int i = a / 7, j = a % 7;
  int h0 = i * 40 / 7, h1 = ((i + 1) * 40 + 6) / 7;
  int w0 = j * 40 / 7, w1 = ((j + 1) * 40 + 6) / 7;
  const float* p = xf + bc * NTOK;
  float s = 0.f; int cnt = 0;
  for (int h = h0; h < h1; ++h)
    for (int w = w0; w < w1; ++w) { s += p[h * 40 + w]; ++cnt; }
  agent[idx] = s / (float)cnt;
}

/* row softmax, runtime cols; launch with blockDim 128/256 (pow2) */
__global__ void softmax_rows(float* __restrict__ x, int cols) {
  long row = blockIdx.x;
  float* p = x + row * (long)cols;
  __shared__ float red[256];
  int tid = threadIdx.x, bs = blockDim.x;
  float mx = -3.0e38f;
  for (int i = tid; i < cols; i += bs) mx = fmaxf(mx, p[i]);
  red[tid] = mx; __syncthreads();
  for (int s = bs >> 1; s > 0; s >>= 1) {
    if (tid < s) red[tid] = fmaxf(red[tid], red[tid + s]);
    __syncthreads();
  }
  mx = red[0]; __syncthreads();
  float sum = 0.f;
  for (int i = tid; i < cols; i += bs) { float e = expf(p[i] - mx); p[i] = e; sum += e; }
  red[tid] = sum; __syncthreads();
  for (int s = bs >> 1; s > 0; s >>= 1) {
    if (tid < s) red[tid] += red[tid + s];
    __syncthreads();
  }
  float inv = 1.f / red[0];
  for (int i = tid; i < cols; i += bs) p[i] *= inv;
}

/* LayerNorm over channels at each (b,n), plus residual:  out = xc + LN(xp) */
__global__ void ln_residual_kernel(const float* __restrict__ xp, const float* __restrict__ xc,
                                   const float* __restrict__ lg, const float* __restrict__ lb,
                                   float* __restrict__ out) {
  long idx = (long)blockIdx.x * 256 + threadIdx.x;
  const long total = (long)BATCH * NTOK;
  if (idx >= total) return;
  int b = (int)(idx / NTOK), n = (int)(idx % NTOK);
  const float* p = xp + (long)b * DCH * NTOK + n;
  float mu = 0.f;
  for (int c = 0; c < DCH; ++c) mu += p[(long)c * NTOK];
  mu *= (1.f / DCH);
  float var = 0.f;
  for (int c = 0; c < DCH; ++c) { float d = p[(long)c * NTOK] - mu; var += d * d; }
  var *= (1.f / DCH);
  float inv = rsqrtf(var + 1e-5f);
  const float* q = xc + (long)b * DCH * NTOK + n;
  float* o = out + (long)b * DCH * NTOK + n;
  for (int c = 0; c < DCH; ++c)
    o[(long)c * NTOK] = q[(long)c * NTOK] + (p[(long)c * NTOK] - mu) * inv * lg[c] + lb[c];
}

/* per-(b,c) unbiased variance of xm = sbuf[:,384:768,:] over HW */
__global__ void __launch_bounds__(256)
chan_var_kernel(const float* __restrict__ sbuf, float* __restrict__ xv) {
  int b = blockIdx.x / DCH, c = blockIdx.x % DCH;
  const float* p = sbuf + ((long)b * (2 * DCH) + DCH + c) * NTOK;
  float s = 0.f, ss = 0.f;
  for (int i = threadIdx.x; i < NTOK; i += 256) { float v = p[i]; s += v; ss += v * v; }
  __shared__ float r1[256], r2[256];
  r1[threadIdx.x] = s; r2[threadIdx.x] = ss; __syncthreads();
  for (int st = 128; st > 0; st >>= 1) {
    if (threadIdx.x < st) { r1[threadIdx.x] += r1[threadIdx.x + st]; r2[threadIdx.x] += r2[threadIdx.x + st]; }
    __syncthreads();
  }
  if (threadIdx.x == 0)
    xv[blockIdx.x] = (r2[0] - r1[0] * r1[0] / (float)NTOK) / (float)(NTOK - 1);
}

/* 8x8 max pool of xm -> [b,384,5,5] */
__global__ void maxpool8_kernel(const float* __restrict__ sbuf, float* __restrict__ xs) {
  long idx = (long)blockIdx.x * 256 + threadIdx.x;
  const long total = (long)BATCH * DCH * 25;
  if (idx >= total) return;
  int ij = (int)(idx % 25); long bc = idx / 25;
  int b = (int)(bc / DCH), c = (int)(bc % DCH);
  int i = ij / 5, j = ij % 5;
  const float* p = sbuf + ((long)b * (2 * DCH) + DCH + c) * NTOK;
  float m = -3.0e38f;
  for (int r = 0; r < 8; ++r)
    for (int q = 0; q < 8; ++q)
      m = fmaxf(m, p[(i * 8 + r) * 40 + j * 8 + q]);
  xs[idx] = m;
}

/* depthwise 3x3 (pad 1) on 5x5 + bias + exact GELU */
__global__ void dw3x3_gelu_kernel(const float* __restrict__ xs, const float* __restrict__ w,
                                  const float* __restrict__ bias, float* __restrict__ out) {
  long idx = (long)blockIdx.x * 256 + threadIdx.x;
  const long total = (long)BATCH * DCH * 25;
  if (idx >= total) return;
  int ij = (int)(idx % 25); long bc = idx / 25;
  int c = (int)(bc % DCH);
  int i = ij / 5, j = ij % 5;
  const float* p = xs + bc * 25;
  float acc = bias[c];
#pragma unroll
  for (int r = 0; r < 3; ++r)
#pragma unroll
    for (int s = 0; s < 3; ++s) {
      int ii = i - 1 + r, jj = j - 1 + s;
      if (ii >= 0 && ii < 5 && jj >= 0 && jj < 5)
        acc += p[ii * 5 + jj] * w[c * 9 + r * 3 + s];
    }
  out[idx] = 0.5f * acc * (1.f + erff(acc * 0.70710678f));
}

/* z = xm * (alpha[c]*xs_up + beta[c]*xv[b,c]) + y */
__global__ void modulate_kernel(const float* __restrict__ sbuf, const float* __restrict__ xsc,
                                const float* __restrict__ xv, const float* __restrict__ alpha,
                                const float* __restrict__ beta, float* __restrict__ z) {
  long idx = (long)blockIdx.x * 256 + threadIdx.x;
  const long total = (long)BATCH * DCH * NTOK;
  if (idx >= total) return;
  int n = (int)(idx % NTOK); long bc = idx / NTOK;
  int b = (int)(bc / DCH), c = (int)(bc % DCH);
  int h = n / 40, w = n % 40;
  float xm = sbuf[((long)b * (2 * DCH) + DCH + c) * NTOK + n];
  float y  = sbuf[((long)b * (2 * DCH) + c) * NTOK + n];
  float xs = xsc[((long)b * DCH + c) * 25 + (h / 8) * 5 + (w / 8)];
  z[idx] = xm * (alpha[c] * xs + beta[c] * xv[b * DCH + c]) + y;
}

/* ---------------- host orchestration ---------------- */

static inline dim3 ggrid(int M, int N, int z) {
  return dim3((N + BN - 1) / BN, (M + BM - 1) / BM, z);
}

extern "C" void kernel_launch(void* const* d_in, const int* in_sizes, int n_in,
                              void* d_out, int out_size, void* d_ws, size_t ws_size,
                              hipStream_t stream) {
  (void)in_sizes; (void)n_in; (void)out_size; (void)ws_size;
  const float* x_p5  = (const float*)d_in[0];
  const float* x_p4  = (const float*)d_in[1];
  const float* x_p3  = (const float*)d_in[2];
  const float* p5_w  = (const float*)d_in[3];
  const float* p5_g  = (const float*)d_in[4];
  const float* p5_b  = (const float*)d_in[5];
  const float* p4_w  = (const float*)d_in[6];
  const float* p4_g  = (const float*)d_in[7];
  const float* p4_b  = (const float*)d_in[8];
  const float* p3_w  = (const float*)d_in[9];
  const float* p3_g  = (const float*)d_in[10];
  const float* p3_b  = (const float*)d_in[11];
  const float* freq_w = (const float*)d_in[12];
  const float* freq_s = (const float*)d_in[13];
  const float* qkv_w  = (const float*)d_in[14];
  const float* proj_w = (const float*)d_in[15];
  const float* proj_b = (const float*)d_in[16];
  const float* ln_g   = (const float*)d_in[17];
  const float* ln_b   = (const float*)d_in[18];
  const float* sm_split_w = (const float*)d_in[19];
  const float* sm_split_b = (const float*)d_in[20];
  const float* sm_dw_w = (const float*)d_in[21];
  const float* sm_dw_b = (const float*)d_in[22];
  const float* sm_pw_w = (const float*)d_in[23];
  const float* sm_pw_b = (const float*)d_in[24];
  const float* sm_alpha = (const float*)d_in[25];
  const float* sm_beta  = (const float*)d_in[26];
  const float* sm_proj_w = (const float*)d_in[27];
  const float* sm_proj_b = (const float*)d_in[28];
  const float* out_w = (const float*)d_in[29];
  const float* out_g = (const float*)d_in[30];
  const float* out_b = (const float*)d_in[31];
  float* dout = (float*)d_out;

  /* workspace carve-out (floats) */
  float* ws = (float*)d_ws;
  size_t off = 0;
  auto alloc = [&](size_t n) { float* p = ws + off; off += n; return p; };
  const long CHW = (long)DCH * NTOK;            /* 614400 */
  float* xc    = alloc((size_t)BATCH * CHW);
  float* xw    = alloc((size_t)BATCH * CHW);
  float* xw2   = alloc((size_t)BATCH * CHW);
  float* xf    = alloc((size_t)BATCH * CHW);
  float* qb    = alloc((size_t)BATCH * CHW);
  float* kb    = alloc((size_t)BATCH * CHW);
  float* vb    = alloc((size_t)BATCH * CHW);
  float* agent = alloc((size_t)BATCH * DCH * AGENTS);
  float* qa    = alloc((size_t)32 * NTOK * AGENTS);
  float* qc    = alloc((size_t)32 * NTOK * HDIM);
  float* attn  = alloc((size_t)NTOK * NTOK);
  float* xo    = alloc((size_t)BATCH * CHW);
  float* xa    = alloc((size_t)BATCH * CHW);
  float* xfu   = alloc((size_t)BATCH * CHW);
  float* sbuf  = alloc((size_t)BATCH * 2 * CHW);
  float* xv    = alloc((size_t)BATCH * DCH);
  float* xs_a  = alloc((size_t)BATCH * DCH * 25);
  float* xs_b  = alloc((size_t)BATCH * DCH * 25);
  float* xs_c  = alloc((size_t)BATCH * DCH * 25);
  float* zb    = alloc((size_t)BATCH * CHW);
  float* xfu2  = alloc((size_t)BATCH * CHW);
  float* p5t   = alloc((size_t)BATCH * HIDC * 400);

  const float attn_scale = 0.10206207261596575f; /* 96^-0.5 */

  /* --- input projections -> xc [b][384][1600] --- */
  gemm_wmma<false, false, 2><<<ggrid(HIDC, 400, BATCH), 128, 0, stream>>>(
      p5_w, x_p5, p5t, HIDC, 400, CIN, CIN, 400, 400,
      0, (long)CIN * 400, (long)HIDC * 400, p5_g, p5_b, nullptr, 0, 1.f);
  resize_bilinear_p5<<<(BATCH * HIDC * NTOK + 255) / 256, 256, 0, stream>>>(p5t, xc);
  gemm_wmma<false, false, 2><<<ggrid(HIDC, NTOK, BATCH), 128, 0, stream>>>(
      p4_w, x_p4, xc + (long)HIDC * NTOK, HIDC, NTOK, CIN, CIN, NTOK, NTOK,
      0, (long)CIN * NTOK, CHW, p4_g, p4_b, nullptr, 0, 1.f);
  gemm_p3_wmma<<<ggrid(HIDC, NTOK, BATCH), 128, 0, stream>>>(
      p3_w, x_p3, xc + (long)2 * HIDC * NTOK, CHW, p3_g, p3_b);

  /* --- wavelet conv --- */
  wav_dec_kernel<<<(BATCH * DCH * 400 + 255) / 256, 256, 0, stream>>>(xc, xw);
  wav_freq_kernel<<<(BATCH * DCH * 4 * 400 + 255) / 256, 256, 0, stream>>>(xw, freq_w, freq_s, xw2);
  wav_rec_kernel<<<(BATCH * DCH * 400 + 255) / 256, 256, 0, stream>>>(xw2, xf);

  /* --- qkv projections (split into q/k/v so (b,h) batching is linear) --- */
  gemm_wmma<false, false, 0><<<ggrid(DCH, NTOK, BATCH), 128, 0, stream>>>(
      qkv_w, xf, qb, DCH, NTOK, DCH, DCH, NTOK, NTOK, 0, CHW, CHW,
      nullptr, nullptr, nullptr, 0, 1.f);
  gemm_wmma<false, false, 0><<<ggrid(DCH, NTOK, BATCH), 128, 0, stream>>>(
      qkv_w + (long)DCH * DCH, xf, kb, DCH, NTOK, DCH, DCH, NTOK, NTOK, 0, CHW, CHW,
      nullptr, nullptr, nullptr, 0, 1.f);
  gemm_wmma<false, false, 0><<<ggrid(DCH, NTOK, BATCH), 128, 0, stream>>>(
      qkv_w + (long)2 * DCH * DCH, xf, vb, DCH, NTOK, DCH, DCH, NTOK, NTOK, 0, CHW, CHW,
      nullptr, nullptr, nullptr, 0, 1.f);

  agent_pool_kernel<<<(BATCH * DCH * AGENTS + 255) / 256, 256, 0, stream>>>(xf, agent);

  /* scores1[z] = scale * q^T(agent):  M=1600 N=49 K=96, TA (q stored [d][n]) */
  gemm_wmma<true, false, 0><<<ggrid(NTOK, AGENTS, 32), 128, 0, stream>>>(
      qb, agent, qa, NTOK, AGENTS, HDIM, NTOK, AGENTS, AGENTS,
      (long)HDIM * NTOK, (long)HDIM * AGENTS, (long)NTOK * AGENTS,
      nullptr, nullptr, nullptr, 0, attn_scale);
  softmax_rows<<<32 * NTOK, 128, 0, stream>>>(qa, AGENTS);
  /* qc[z] = qa * agent^T : M=1600 N=96 K=49, TB (agent stored [d][a]) */
  gemm_wmma<false, true, 0><<<ggrid(NTOK, HDIM, 32), 128, 0, stream>>>(
      qa, agent, qc, NTOK, HDIM, AGENTS, AGENTS, AGENTS, HDIM,
      (long)NTOK * AGENTS, (long)HDIM * AGENTS, (long)NTOK * HDIM,
      nullptr, nullptr, nullptr, 0, 1.f);

  /* per-(b,h): attn = softmax(scale * qc k) ; xo^T = v attn^T  (reuse attn buf) */
  for (int z = 0; z < 32; ++z) {
    const long zo = (long)z * HDIM * NTOK;
    gemm_wmma<false, false, 0><<<ggrid(NTOK, NTOK, 1), 128, 0, stream>>>(
        qc + (long)z * NTOK * HDIM, kb + zo, attn, NTOK, NTOK, HDIM,
        HDIM, NTOK, NTOK, 0, 0, 0, nullptr, nullptr, nullptr, 0, attn_scale);
    softmax_rows<<<NTOK, 256, 0, stream>>>(attn, NTOK);
    /* xo[d][n] = sum_m v[d][m] * attn[n][m] : M=96 N=1600 K=1600, TB */
    gemm_wmma<false, true, 0><<<ggrid(HDIM, NTOK, 1), 128, 0, stream>>>(
        vb + zo, attn, xo + zo, HDIM, NTOK, NTOK,
        NTOK, NTOK, NTOK, 0, 0, 0, nullptr, nullptr, nullptr, 0, 1.f);
  }

  /* proj + bias, then LN over channels + residual add with xc -> xfu */
  gemm_wmma<false, false, 1><<<ggrid(DCH, NTOK, BATCH), 128, 0, stream>>>(
      proj_w, xo, xa, DCH, NTOK, DCH, DCH, NTOK, NTOK, 0, CHW, CHW,
      nullptr, proj_b, nullptr, 0, 1.f);
  ln_residual_kernel<<<(BATCH * NTOK + 255) / 256, 256, 0, stream>>>(xa, xc, ln_g, ln_b, xfu);

  /* --- stat_mod --- */
  gemm_wmma<false, false, 1><<<ggrid(2 * DCH, NTOK, BATCH), 128, 0, stream>>>(
      sm_split_w, xfu, sbuf, 2 * DCH, NTOK, DCH, DCH, NTOK, NTOK, 0, CHW, 2 * CHW,
      nullptr, sm_split_b, nullptr, 0, 1.f);
  chan_var_kernel<<<BATCH * DCH, 256, 0, stream>>>(sbuf, xv);
  maxpool8_kernel<<<(BATCH * DCH * 25 + 255) / 256, 256, 0, stream>>>(sbuf, xs_a);
  dw3x3_gelu_kernel<<<(BATCH * DCH * 25 + 255) / 256, 256, 0, stream>>>(xs_a, sm_dw_w, sm_dw_b, xs_b);
  gemm_wmma<false, false, 1><<<ggrid(DCH, 25, BATCH), 128, 0, stream>>>(
      sm_pw_w, xs_b, xs_c, DCH, 25, DCH, DCH, 25, 25,
      0, (long)DCH * 25, (long)DCH * 25, nullptr, sm_pw_b, nullptr, 0, 1.f);
  modulate_kernel<<<(BATCH * DCH * NTOK + 255) / 256, 256, 0, stream>>>(
      sbuf, xs_c, xv, sm_alpha, sm_beta, zb);
  /* sm_proj + bias + residual(xfu) -> xfu2 */
  gemm_wmma<false, false, 3><<<ggrid(DCH, NTOK, BATCH), 128, 0, stream>>>(
      sm_proj_w, zb, xfu2, DCH, NTOK, DCH, DCH, NTOK, NTOK, 0, CHW, CHW,
      nullptr, sm_proj_b, xfu, CHW, 1.f);

  /* --- output conv + BN + SiLU -> d_out --- */
  gemm_wmma<false, false, 2><<<ggrid(256, NTOK, BATCH), 128, 0, stream>>>(
      out_w, xfu2, dout, 256, NTOK, DCH, DCH, NTOK, NTOK, 0, CHW, (long)256 * NTOK,
      out_g, out_b, nullptr, 0, 1.f);
}
